// HierarchicalAggregate_72138270703838
// MI455X (gfx1250) — matrix operand order repaired and checked by reading, not compiled, not run on hardware
//
#include <hip/hip_runtime.h>
#include <hip/hip_bf16.h>

// ---------------- problem constants (match reference) ----------------
#define N_CONCEPTS 100000
#define D_FEAT     256
#define BATCH      1024

// LDS B-tile layout: row data = 512 B (256 bf16), padded to 528 B
// (33 x 16 B chunks) -> 132-dword row stride == 4-bank rotation/row,
// so the 16 lanes of a half-wave hit disjoint bank groups.
#define ROW_BYTES   528u
#define TILE_BYTES  (32u * ROW_BYTES)      // 32 rows (2 N-tiles) = 16896 B
#define BUF_BYTES   (2u * TILE_BYTES)      // hi + lo regions     = 33792 B

typedef __attribute__((ext_vector_type(16))) __bf16 v16bf;
typedef __attribute__((ext_vector_type(8)))  float  v8f;

union FragBF16 {            // one WMMA 16-bit operand: 16 bf16 per lane = 32 B
    v16bf v;
    uint4 q[2];
};

// ---------------------------------------------------------------------
// Kernel 1: scatter  aw[rows[e],:] += values[e] * w[cols[e],:]
// One wave32 per edge; each lane handles 8 f32 (2 x float4, 128 floats apart).
// w (102.4 MB) is L2-resident; atomics go to the L2 atomic units.
// ---------------------------------------------------------------------
__global__ __launch_bounds__(256)
void ha_scatter_kernel(const float* __restrict__ w,
                       const float* __restrict__ values,
                       const int*   __restrict__ rows,
                       const int*   __restrict__ cols,
                       float*       __restrict__ aw,
                       int nnz)
{
    const int lane = threadIdx.x & 31;
    const int edge = blockIdx.x * (blockDim.x >> 5) + (threadIdx.x >> 5);
    if (edge >= nnz) return;

    const int   r = rows[edge];
    const int   c = cols[edge];
    const float v = values[edge];

    const float4* __restrict__ wsrc = (const float4*)(w + (size_t)c * D_FEAT);
    float* __restrict__ dst = aw + (size_t)r * D_FEAT;

#pragma unroll
    for (int i = 0; i < 2; ++i) {
        const int qi = lane + 32 * i;          // float4 index within the row
        float4 t = wsrc[qi];
        const int base = qi * 4;
        atomicAdd(dst + base + 0, v * t.x);
        atomicAdd(dst + base + 1, v * t.y);
        atomicAdd(dst + base + 2, v * t.z);
        atomicAdd(dst + base + 3, v * t.w);
    }
}

// ---------------------------------------------------------------------
// Kernel 2: f32 -> (bf16 hi, bf16 lo) split.  x ~= hi + lo with ~16
// mantissa bits total; GEMM keeps hi*hi + hi*lo + lo*hi.
// ---------------------------------------------------------------------
__global__ __launch_bounds__(256)
void ha_split_bf16_kernel(const float* __restrict__ src,
                          __bf16* __restrict__ hi,
                          __bf16* __restrict__ lo,
                          long long n)
{
    long long i = (long long)blockIdx.x * blockDim.x + threadIdx.x;
    if (i >= n) return;
    float  x = src[i];
    __bf16 h = (__bf16)x;
    float  hf = (float)h;
    __bf16 l = (__bf16)(x - hf);
    hi[i] = h;
    lo[i] = l;
}

// ---------------------------------------------------------------------
// Kernel 3: WMMA GEMM  out[B=1024, N=100000] = inputs @ aw^T + b
// 8 waves / block; wave w owns M-tile (blockIdx.x*8 + w)*16; A (hi+lo,
// full K=256) is held in VGPRs for the whole block sweep.
// Block sweeps 5 N-tile PAIRS; each 33 KB pair (32 aw rows x 256 K x
// bf16 hi+lo) is staged into bank-conflict-padded LDS with
// global_load_async_to_lds_b128 (ASYNCcnt), double-buffered so the
// async fill of pair p+1 overlaps the 48 WMMAs of pair p.
// Each wave keeps TWO accumulators (one per tile of the pair): tile-0
// WMMAs cover tile-1 fragment LDS latency and vice versa, so the
// compiler's s_wait_dscnt sits behind independent XDL work.
//
// Fragment addressing (ISA 7.12.2, wave32):
//   A 16-bit 16x32: lane L<16 -> row M=L, K=0..7 & 16..23; lanes>=16: +8.
//   B 16-bit 32x16: lane L -> column N=L&15; lanes 0-15: K=0..7 & 16..23,
//                   lanes 16-31: K=8..15 & 24..31 (2 x 16 B, 32 B apart).
//   C/D f32 16x16:  reg r, lane L -> M = r + (L>=16 ? 8 : 0), N = L&15.
// ---------------------------------------------------------------------
__global__ __launch_bounds__(256)
void ha_gemm_wmma_kernel(const __bf16* __restrict__ in_hi,
                         const __bf16* __restrict__ in_lo,
                         const __bf16* __restrict__ aw_hi,
                         const __bf16* __restrict__ aw_lo,
                         const float*  __restrict__ bias,
                         float*        __restrict__ out)
{
    __shared__ uint4 sB[2 * BUF_BYTES / 16];   // 67,584 B total

    const int tid   = threadIdx.x;
    const int wave  = tid >> 5;
    const int lane  = tid & 31;
    const int lhalf = lane >> 4;       // 0: lanes 0-15, 1: lanes 16-31
    const int lidx  = lane & 15;

    const int mbase = (blockIdx.x * 8 + wave) * 16;
    const int am    = mbase + lidx;    // A row for this lane

    // ---- preload A fragments (full K, hi+lo) into VGPRs ----
    FragBF16 aHi[8], aLo[8];
    {
        const __bf16* __restrict__ arow_hi = in_hi + (size_t)am * D_FEAT;
        const __bf16* __restrict__ arow_lo = in_lo + (size_t)am * D_FEAT;
#pragma unroll
        for (int kc = 0; kc < 8; ++kc) {
            const int kA = kc * 32 + lhalf * 8;
            aHi[kc].q[0] = *(const uint4*)(arow_hi + kA);
            aHi[kc].q[1] = *(const uint4*)(arow_hi + kA + 16);
            aLo[kc].q[0] = *(const uint4*)(arow_lo + kA);
            aLo[kc].q[1] = *(const uint4*)(arow_lo + kA + 16);
        }
    }

    const uint32_t ldsBase = (uint32_t)(uintptr_t)(&sB[0]);

    // async-stage N-tile pair p (32 aw rows, hi+lo) into LDS buffer b:
    // 8 b128 async loads / thread, scattered into the padded row layout.
    auto issuePair = [&](int p, int b) {
        const size_t rowbase = (size_t)((blockIdx.y * 10 + 2 * p) * 16) * D_FEAT;
        const unsigned long long gHi = (unsigned long long)(aw_hi + rowbase);
        const unsigned long long gLo = (unsigned long long)(aw_lo + rowbase);
        const uint32_t bufOff = ldsBase + (uint32_t)b * BUF_BYTES;
#pragma unroll
        for (int j = 0; j < 4; ++j) {
            const uint32_t ci   = (uint32_t)(tid + 256 * j);   // 0..1023
            const uint32_t voff = ci * 16u;                    // linear global
            const uint32_t row  = ci >> 5;                     // 0..31
            const uint32_t cno  = ci & 31u;                    // chunk in row
            const uint32_t d    = bufOff + row * ROW_BYTES + cno * 16u;
            asm volatile("global_load_async_to_lds_b128 %0, %1, %2"
                         :: "v"(d), "v"(voff), "s"(gHi) : "memory");
            asm volatile("global_load_async_to_lds_b128 %0, %1, %2"
                         :: "v"(d + TILE_BYTES), "v"(voff), "s"(gLo) : "memory");
        }
    };

    issuePair(0, 0);
    issuePair(1, 1);

    for (int p = 0; p < 5; ++p) {
        // pair p's 8 async loads are the oldest; <=8 outstanding => landed
        if (p < 4) asm volatile("s_wait_asynccnt 0x8" ::: "memory");
        else       asm volatile("s_wait_asynccnt 0x0" ::: "memory");
        __syncthreads();

        const int buf = p & 1;
        const char* hiBase = (const char*)(&sB[0]) + buf * BUF_BYTES;
        const char* loBase = hiBase + TILE_BYTES;

        auto loadFrag = [&](FragBF16& fh, FragBF16& fl, int row, int kc) {
            const int boff = row * (int)ROW_BYTES + (kc * 4 + lhalf) * 16;
            fh.q[0] = *(const uint4*)(hiBase + boff);
            fh.q[1] = *(const uint4*)(hiBase + boff + 32);
            fl.q[0] = *(const uint4*)(loBase + boff);
            fl.q[1] = *(const uint4*)(loBase + boff + 32);
        };

        const int row0 = lidx;          // tile 0 of the pair
        const int row1 = 16 + lidx;     // tile 1 of the pair

        v8f acc0 = {}, acc1 = {};
        FragBF16 f0h, f0l, f1h, f1l;
        loadFrag(f0h, f0l, row0, 0);

#pragma unroll
        for (int kc = 0; kc < 8; ++kc) {
            loadFrag(f1h, f1l, row1, kc);           // stream-1 fragment

            acc0 = __builtin_amdgcn_wmma_f32_16x16x32_bf16(
                       false, aHi[kc].v, false, f0h.v, (short)0, acc0, false, false);
            acc0 = __builtin_amdgcn_wmma_f32_16x16x32_bf16(
                       false, aLo[kc].v, false, f0h.v, (short)0, acc0, false, false);
            acc0 = __builtin_amdgcn_wmma_f32_16x16x32_bf16(
                       false, aHi[kc].v, false, f0l.v, (short)0, acc0, false, false);

            if (kc < 7) loadFrag(f0h, f0l, row0, kc + 1);   // stream-0 prefetch

            acc1 = __builtin_amdgcn_wmma_f32_16x16x32_bf16(
                       false, aHi[kc].v, false, f1h.v, (short)0, acc1, false, false);
            acc1 = __builtin_amdgcn_wmma_f32_16x16x32_bf16(
                       false, aLo[kc].v, false, f1h.v, (short)0, acc1, false, false);
            acc1 = __builtin_amdgcn_wmma_f32_16x16x32_bf16(
                       false, aHi[kc].v, false, f1l.v, (short)0, acc1, false, false);
        }

        const int   nbase0 = (blockIdx.y * 10 + 2 * p) * 16;
        const int   nbase1 = nbase0 + 16;
        const float bv0    = bias[nbase0 + lidx];
        const float bv1    = bias[nbase1 + lidx];
#pragma unroll
        for (int r = 0; r < 8; ++r) {
            const int row = mbase + r + lhalf * 8;
            out[(size_t)row * N_CONCEPTS + nbase0 + lidx] = acc0[r] + bv0;
            out[(size_t)row * N_CONCEPTS + nbase1 + lidx] = acc1[r] + bv1;
        }

        __syncthreads();                 // all waves done reading buf
        if (p + 2 <= 4) issuePair(p + 2, buf);
    }
}

// ---------------------------------------------------------------------
// Host-side launch.  Inputs (setup_inputs order):
//   d_in[0]=inputs[1024,256] f32, d_in[1]=w[100000,256] f32,
//   d_in[2]=b[100000] f32, d_in[3]=values[2M] f32,
//   d_in[4]=rows[2M] i32, d_in[5]=cols[2M] i32.
// Workspace layout (256-B aligned):
//   aw_f32 102.4MB | aw_hi 51.2MB | aw_lo 51.2MB | in_hi 0.5MB | in_lo 0.5MB
// ---------------------------------------------------------------------
extern "C" void kernel_launch(void* const* d_in, const int* in_sizes, int n_in,
                              void* d_out, int out_size, void* d_ws, size_t ws_size,
                              hipStream_t stream)
{
    const float* inputs = (const float*)d_in[0];
    const float* w      = (const float*)d_in[1];
    const float* b      = (const float*)d_in[2];
    const float* values = (const float*)d_in[3];
    const int*   rows   = (const int*)d_in[4];
    const int*   cols   = (const int*)d_in[5];
    float*       out    = (float*)d_out;

    const int nnz = in_sizes[3];

    const size_t awElems = (size_t)N_CONCEPTS * D_FEAT;   // 25,600,000
    const size_t inElems = (size_t)BATCH * D_FEAT;        //    262,144

    char* ws = (char*)d_ws;
    float*  aw    = (float*)(ws);
    __bf16* aw_hi = (__bf16*)(ws + awElems * sizeof(float));
    __bf16* aw_lo = (__bf16*)(ws + awElems * sizeof(float) + awElems * sizeof(__bf16));
    __bf16* in_hi = (__bf16*)(ws + awElems * sizeof(float) + 2 * awElems * sizeof(__bf16));
    __bf16* in_lo = (__bf16*)(ws + awElems * sizeof(float) + 2 * awElems * sizeof(__bf16)
                                 + inElems * sizeof(__bf16));

    // 1) zero the segment-sum accumulator
    hipMemsetAsync(aw, 0, awElems * sizeof(float), stream);

    // 2) scatter: one wave per edge (8 waves / 256-thread block)
    {
        const int wavesPerBlock = 256 / 32;
        const int grid = (nnz + wavesPerBlock - 1) / wavesPerBlock;
        ha_scatter_kernel<<<grid, 256, 0, stream>>>(w, values, rows, cols, aw, nnz);
    }

    // 3) split f32 -> bf16 hi/lo (aw and inputs)
    {
        const long long nA = (long long)awElems;
        ha_split_bf16_kernel<<<(int)((nA + 255) / 256), 256, 0, stream>>>(aw, aw_hi, aw_lo, nA);
        const long long nI = (long long)inElems;
        ha_split_bf16_kernel<<<(int)((nI + 255) / 256), 256, 0, stream>>>(inputs, in_hi, in_lo, nI);
    }

    // 4) WMMA GEMM: grid = (M tiles / 8 waves, N tile-pair-groups of 5)
    {
        dim3 grid(BATCH / (8 * 16), (N_CONCEPTS / 16) / 10);   // (8, 625)
        ha_gemm_wmma_kernel<<<grid, 256, 0, stream>>>(in_hi, in_lo, aw_hi, aw_lo, b, out);
    }
}
